// RFCN_71940702208530
// MI455X (gfx1250) — matrix operand order repaired and checked by reading, not compiled, not run on hardware
//
#include <hip/hip_runtime.h>
#include <hip/hip_bf16.h>
#include <stdint.h>

// R-FCN head for MI455X (gfx1250, wave32, WMMA + TDM).
// Restructure: x = feat @ w_new^T  -> PSRoI-pool 512-d x features ->
// per-bin 16x16-tile WMMA head GEMMs (bias folded, 1/49 mean via f32 atomics).
//
// Workspace (~30.1 MB): x16 f16 [B][H*W][512] then pooled f16 [R][49][512].

#define PSn 7
#define NSn 2
#define NCLS 81
#define Bn 2
#define CINn 1024
#define Hfn 38
#define Wfn 56
#define Sfn (Hfn * Wfn)   /* 2128 = 133*16 */
#define FEATn 512
#define Rn 512
#define NBIN 49

typedef __attribute__((ext_vector_type(16))) _Float16 v16h;
typedef __attribute__((ext_vector_type(8)))  _Float16 v8h;
typedef __attribute__((ext_vector_type(2)))  _Float16 h2;
typedef __attribute__((ext_vector_type(8)))  float    v8f;
typedef __attribute__((ext_vector_type(4)))  unsigned int v4u;
typedef __attribute__((ext_vector_type(8)))  int      v8i;
typedef __attribute__((ext_vector_type(4)))  int      v4i;

#ifndef __has_builtin
#define __has_builtin(x) 0
#endif
#if __has_builtin(__builtin_amdgcn_tensor_load_to_lds)
#define HAVE_TDM 1
#else
#define HAVE_TDM 0
#endif

#if HAVE_TDM
// Issue one TDM 2D tile load: 16 (dim0, contiguous) x 32 (dim1, stride Sfn)
// f32 elements, global -> LDS at lds_dst. D# per ISA §8.3/8.4.
// lds_addr is taken from the generic pointer's low 32 bits (flat->LDS aperture
// mapping); the ptrtoint also escapes the LDS object so
// the compiler cannot fold the subsequent ds_load reads of the tile to undef.
__device__ __forceinline__ void tdm_stage(const float* gsrc, const float* lds_dst) {
  const unsigned long long ga = (unsigned long long)(uintptr_t)gsrc;
  const unsigned lds_off = (unsigned)(uintptr_t)lds_dst;  // LDS byte offset
  v4u g0;
  g0[0] = 1u;                                          // count=1 (valid), user mode
  g0[1] = lds_off;                                     // lds_addr (bytes)
  g0[2] = (unsigned)(ga & 0xffffffffull);              // global_addr[31:0]
  g0[3] = (unsigned)((ga >> 32) & 0x1ffffffull)        // global_addr[56:32]
        | (2u << 30);                                  // type=2 ("image")
  const unsigned td0 = 0x7fffffffu, td1 = 0x7fffffffu; // huge dims: no OOB clip
  v8i g1;
  g1[0] = (int)(2u << 16);                             // wg_mask=0, data_size=2 (4B)
  g1[1] = (int)((td0 & 0xffffu) << 16);                // abar=0, tensor_dim0 lo
  g1[2] = (int)(((td0 >> 16) & 0xffffu) | ((td1 & 0xffffu) << 16));
  g1[3] = (int)(((td1 >> 16) & 0xffffu) | (16u << 16));// tile_dim0 = 16
  g1[4] = (int)32u;                                    // tile_dim1 = 32, tile_dim2=0
  g1[5] = (int)(unsigned)Sfn;                          // tensor_dim0_stride[31:0]
  g1[6] = 0;                                           // stride hi / dim1_stride lo
  g1[7] = 0;
  const v4i z4 = {0, 0, 0, 0};                         // groups 2/3 unused (2D)
  const v8i z8 = {0, 0, 0, 0, 0, 0, 0, 0};
  __builtin_amdgcn_tensor_load_to_lds(g0, g1, z4, z4, z8, 0);
}
#endif

__device__ __forceinline__ void stage_fallback(float* dst, const float* src, int tid) {
  for (int e = tid; e < 32 * 16; e += 256) {
    const int c = e >> 4, s = e & 15;
    dst[e] = src[(size_t)c * Sfn + s];
  }
}

// ---------------------------------------------------------------------------
// K1: x16[b][s][o] = sum_c feat[b][c][s] * w_new[o][c]   (f32 accum, f16 out)
// Block = 8 waves; macro-tile 16(s) x 256(o); wave w owns o = og*256+w*32..+31.
// feat K-panels staged to LDS by TDM (double-buffered, s_wait_tensorcnt);
// A fragments (w_new, M=o) are contiguous global b128 loads; D packs to one
// 16B f16 store per fragment per lane.
// ---------------------------------------------------------------------------
__global__ __launch_bounds__(256) void k_gemm_x(const float* __restrict__ feat,
                                                const float* __restrict__ w_new,
                                                _Float16* __restrict__ x16) {
  __shared__ float tile[2][32 * 16];                   // 4 KB double buffer
  const int st = blockIdx.x;                           // s tile 0..132
  const int og = blockIdx.y;                           // o group 0..1
  const int b  = blockIdx.z;
  const int tid = threadIdx.x;
  const int l = tid & 31, lh = l >> 4, ll = l & 15;
  const int s0 = st * 16;
  const int o0 = og * 256 + (tid >> 5) * 32;
  const float* fbase = feat + (size_t)b * CINn * Sfn + s0;
  // Scalar (SALU) branch gate: TDM ignores EXEC, so only wave 0 may reach it.
  const int wave_first_tid = __builtin_amdgcn_readfirstlane((int)threadIdx.x);

  const int NK = CINn / 32;
  v8f acc0 = {}, acc1 = {};
#if HAVE_TDM
  if (wave_first_tid == 0) tdm_stage(fbase, &tile[0][0]);
#else
  stage_fallback(tile[0], fbase, tid);
#endif
  for (int kt = 0; kt < NK; ++kt) {
    const int kb = kt * 32;
#if HAVE_TDM
    if (wave_first_tid == 0) {
      if (kt + 1 < NK) {
        tdm_stage(fbase + (size_t)(kt + 1) * 32 * Sfn, &tile[(kt + 1) & 1][0]);
        __builtin_amdgcn_s_wait_tensorcnt(1);          // current buffer ready
      } else {
        __builtin_amdgcn_s_wait_tensorcnt(0);
      }
      asm volatile("" ::: "memory");                   // TDM wrote LDS: no reorder
    }
#else
    if (kt + 1 < NK)
      stage_fallback(tile[(kt + 1) & 1], fbase + (size_t)(kt + 1) * 32 * Sfn, tid);
#endif
    __syncthreads();                                   // buffer ready for all waves
    const float* tl = tile[kt & 1];

    // B fragment (feat tile, KxN = c x s): lane -> N = ll; elems j<8 -> K=8lh+j, j>=8 -> K=16+8lh+(j-8).
    v16h bfr;
#pragma unroll
    for (int j = 0; j < 8; ++j) {
      bfr[j]     = (_Float16)tl[(8 * lh + j) * 16 + ll];
      bfr[j + 8] = (_Float16)tl[(16 + 8 * lh + j) * 16 + ll];
    }

    // A fragments (w_new, MxK = o x c): lane -> M = ll; elems j<8 -> K=kb+8lh+j,
    // j>=8 -> K=kb+16+8lh+(j-8): two contiguous 8-float runs -> b128 loads.
    const float* w0 = w_new + (size_t)(o0 + ll) * CINn + kb;
    const float* w1 = w_new + (size_t)(o0 + 16 + ll) * CINn + kb;
    v16h a0, a1;
#pragma unroll
    for (int j = 0; j < 8; ++j) {
      a0[j]     = (_Float16)w0[8 * lh + j];
      a0[j + 8] = (_Float16)w0[16 + 8 * lh + j];
      a1[j]     = (_Float16)w1[8 * lh + j];
      a1[j + 8] = (_Float16)w1[16 + 8 * lh + j];
    }
    acc0 = __builtin_amdgcn_wmma_f32_16x16x32_f16(false, a0, false, bfr,
                                                  (short)0, acc0, false, false);
    acc1 = __builtin_amdgcn_wmma_f32_16x16x32_f16(false, a1, false, bfr,
                                                  (short)0, acc1, false, false);
    __syncthreads();                                   // reads done before reuse
  }
  // D: M(=o) = i + 8*lh, N(=s) = ll -> 8 consecutive o per lane: packed store.
  v8h p0, p1;
#pragma unroll
  for (int i = 0; i < 8; ++i) {
    p0[i] = (_Float16)acc0[i];
    p1[i] = (_Float16)acc1[i];
  }
  _Float16* xr = x16 + ((size_t)b * Sfn + s0 + ll) * FEATn;
  *(v8h*)(xr + o0 + 8 * lh)      = p0;
  *(v8h*)(xr + o0 + 16 + 8 * lh) = p1;
}

// ---------------------------------------------------------------------------
// K2: PSRoI-align pooling of the 512-d x features, channel pairs per thread.
// pooled[r][bin][o] = mean over 2x2 samples of bilinear(x16[b], y, x, o).
// ---------------------------------------------------------------------------
__global__ __launch_bounds__(256) void k_pool(const float* __restrict__ rois,
                                              const _Float16* __restrict__ x16,
                                              _Float16* __restrict__ pooled,
                                              const int* __restrict__ stride_p) {
  const int bin = blockIdx.x;
  const int r   = blockIdx.y;
  const int ph = bin / PSn, pw = bin % PSn;
  const float inv = 1.0f / (float)stride_p[0];
  const float* roi = rois + (size_t)r * 5;
  const int   bi = (int)roi[0];
  const float x1 = roi[1] * inv, y1 = roi[2] * inv;
  const float x2 = roi[3] * inv, y2 = roi[4] * inv;
  const float bw = fmaxf(x2 - x1, 0.1f) * (1.0f / PSn);
  const float bh = fmaxf(y2 - y1, 0.1f) * (1.0f / PSn);

  int   iy0[NSn], iy1[NSn], ix0[NSn], ix1[NSn];
  float wy[NSn], wx[NSn];
#pragma unroll
  for (int i = 0; i < NSn; ++i) {
    const float off = ((float)i + 0.5f) / (float)NSn;
    float yy = y1 + ((float)ph + off) * bh;
    yy = fminf(fmaxf(yy, 0.0f), (float)(Hfn - 1));
    const float y0f = floorf(yy);
    iy0[i] = (int)y0f;
    iy1[i] = min(iy0[i] + 1, Hfn - 1);
    wy[i]  = yy - y0f;
    float xx = x1 + ((float)pw + off) * bw;
    xx = fminf(fmaxf(xx, 0.0f), (float)(Wfn - 1));
    const float x0f = floorf(xx);
    ix0[i] = (int)x0f;
    ix1[i] = min(ix0[i] + 1, Wfn - 1);
    wx[i]  = xx - x0f;
  }

  const size_t base = (size_t)bi * Sfn * FEATn;
  const int o = 2 * (int)threadIdx.x;                 // channel pair
  float ax = 0.0f, ay = 0.0f;
#pragma unroll
  for (int j = 0; j < NSn; ++j) {
#pragma unroll
    for (int i = 0; i < NSn; ++i) {
      const float w00 = (1.0f - wy[j]) * (1.0f - wx[i]);
      const float w01 = (1.0f - wy[j]) * wx[i];
      const float w10 = wy[j] * (1.0f - wx[i]);
      const float w11 = wy[j] * wx[i];
      const h2 v00 = *(const h2*)(x16 + base + (size_t)(iy0[j] * Wfn + ix0[i]) * FEATn + o);
      const h2 v01 = *(const h2*)(x16 + base + (size_t)(iy0[j] * Wfn + ix1[i]) * FEATn + o);
      const h2 v10 = *(const h2*)(x16 + base + (size_t)(iy1[j] * Wfn + ix0[i]) * FEATn + o);
      const h2 v11 = *(const h2*)(x16 + base + (size_t)(iy1[j] * Wfn + ix1[i]) * FEATn + o);
      ax += w00 * (float)v00[0] + w01 * (float)v01[0] + w10 * (float)v10[0] + w11 * (float)v11[0];
      ay += w00 * (float)v00[1] + w01 * (float)v01[1] + w10 * (float)v10[1] + w11 * (float)v11[1];
    }
  }
  h2 res;
  res[0] = (_Float16)(ax * (1.0f / (NSn * NSn)));
  res[1] = (_Float16)(ay * (1.0f / (NSn * NSn)));
  *(h2*)(pooled + ((size_t)r * NBIN + bin) * FEATn + o) = res;
}

// ---------------------------------------------------------------------------
// K3a: outputs start as bin-averaged biases (bilinear weights sum to 1).
// ---------------------------------------------------------------------------
__global__ void k_init_out(const float* __restrict__ b_score,
                           const float* __restrict__ b_bbox,
                           float* __restrict__ out) {
  const int idx = blockIdx.x * blockDim.x + threadIdx.x;
  const int ncls = Rn * NCLS;
  if (idx < ncls) {
    const int c = idx % NCLS;
    float s = 0.0f;
    for (int bin = 0; bin < NBIN; ++bin) s += b_score[c * NBIN + bin];
    out[idx] = s * (1.0f / NBIN);
  } else if (idx < ncls + Rn * 4) {
    const int c = (idx - ncls) % 4;
    float s = 0.0f;
    for (int bin = 0; bin < NBIN; ++bin) s += b_bbox[c * NBIN + bin];
    out[idx] = s * (1.0f / NBIN);
  }
}

// ---------------------------------------------------------------------------
// K3b: per-bin head GEMM, out[r][c] += (1/49) * pooled[r][bin][:] . w[c*49+bin][:]
// A fragment = two aligned 16B f16 loads per lane; f32 atomics across bins.
// ---------------------------------------------------------------------------
__global__ __launch_bounds__(32) void k_head(const _Float16* __restrict__ pooled,
                                             const float* __restrict__ w,
                                             float* __restrict__ out, int ncols) {
  const int bin = blockIdx.x;
  const int ct  = blockIdx.y;
  const int rt  = blockIdx.z;
  const int l = threadIdx.x, lh = l >> 4, ll = l & 15;
  const int r0 = rt * 16, c0 = ct * 16;
  const int c_lane = min(c0 + ll, ncols - 1);          // clamp: no OOB w reads
  const _Float16* Arow = pooled + ((size_t)(r0 + ll) * NBIN + bin) * FEATn;
  const float*    wrow = w + ((size_t)c_lane * NBIN + bin) * FEATn;

  v8f acc = {};
  for (int kt = 0; kt < FEATn / 32; ++kt) {
    const int kb = kt * 32;
    const v8h alo = *(const v8h*)(Arow + kb + 8 * lh);        // K = kb+8lh+0..7
    const v8h ahi = *(const v8h*)(Arow + kb + 16 + 8 * lh);   // K = kb+16+8lh+0..7
    v16h a;
#pragma unroll
    for (int j = 0; j < 8; ++j) { a[j] = alo[j]; a[j + 8] = ahi[j]; }
    v16h bf;
#pragma unroll
    for (int j = 0; j < 8; ++j) {
      bf[j]     = (_Float16)wrow[kb + 8 * lh + j];            // K = kb+8lh+j
      bf[j + 8] = (_Float16)wrow[kb + 16 + 8 * lh + j];       // K = kb+16+8lh+j
    }
    acc = __builtin_amdgcn_wmma_f32_16x16x32_f16(false, a, false, bf,
                                                 (short)0, acc, false, false);
  }
  const float sc = 1.0f / (float)NBIN;
#pragma unroll
  for (int i = 0; i < 8; ++i) {
    const int rr = r0 + i + 8 * lh;
    const int cc = c0 + ll;
    if (cc < ncols) atomicAdd(&out[(size_t)rr * ncols + cc], acc[i] * sc);
  }
}

// ---------------------------------------------------------------------------
extern "C" void kernel_launch(void* const* d_in, const int* in_sizes, int n_in,
                              void* d_out, int out_size, void* d_ws, size_t ws_size,
                              hipStream_t stream) {
  (void)in_sizes; (void)n_in; (void)out_size; (void)ws_size;
  const float* rois    = (const float*)d_in[0];
  const float* feat    = (const float*)d_in[1];
  const float* w_new   = (const float*)d_in[2];
  const float* w_score = (const float*)d_in[3];
  const float* b_score = (const float*)d_in[4];
  const float* w_bbox  = (const float*)d_in[5];
  const float* b_bbox  = (const float*)d_in[6];
  const int*   stridep = (const int*)d_in[7];
  float* out = (float*)d_out;

  const size_t x16_bytes = (size_t)Bn * Sfn * FEATn * sizeof(_Float16);
  _Float16* x16    = (_Float16*)d_ws;
  _Float16* pooled = (_Float16*)((char*)d_ws + ((x16_bytes + 255) & ~(size_t)255));

  // 1) x = feat @ w_new^T : 532 blocks x 8 waves, TDM-staged K panels.
  k_gemm_x<<<dim3(Sfn / 16, 2, Bn), 256, 0, stream>>>(feat, w_new, x16);
  // 2) PSRoI pooling of x : one block per (roi, bin).
  k_pool<<<dim3(NBIN, Rn), 256, 0, stream>>>(rois, x16, pooled, stridep);
  // 3) bias init, then per-bin head GEMMs accumulated with f32 atomics.
  k_init_out<<<(Rn * (NCLS + 4) + 255) / 256, 256, 0, stream>>>(b_score, b_bbox, out);
  k_head<<<dim3(NBIN, (NCLS + 15) / 16, Rn / 16), 32, 0, stream>>>(
      pooled, w_score, out, NCLS);
  k_head<<<dim3(NBIN, 1, Rn / 16), 32, 0, stream>>>(
      pooled, w_bbox, out + (size_t)Rn * NCLS, 4);
}
